// SelectiveScan_69269232550328
// MI455X (gfx1250) — matrix operand order repaired and checked by reading, not compiled, not run on hardware
//
#include <hip/hip_runtime.h>

// ---------------- problem constants (from the reference) ----------------
#define BATCH    2
#define SEQ      2048
#define D_MODEL  1024
#define D_INNER  2048
#define DT_RANK  64
#define D_STATE  16
#define CONV_K   4
#define NTOK     (BATCH * SEQ)           // 4096 tokens
#define XZ_N     (2 * D_INNER)           // 4096
#define XDBL_N   (DT_RANK + 2 * D_STATE) // 96

// ---------------- WMMA / TDM types ----------------
typedef __attribute__((ext_vector_type(16))) __bf16        v16bf;
typedef __attribute__((ext_vector_type(8)))  float         v8f;
typedef __attribute__((ext_vector_type(4)))  unsigned int  v4u;
typedef __attribute__((ext_vector_type(8)))  int           v8i_t;
typedef __attribute__((ext_vector_type(4)))  int           v4i_t;

struct U4 { unsigned int x, y, z, w; };
union alignas(32) Frag {
  U4    u[2];
  v16bf v;
};

__device__ __forceinline__ unsigned short f32_to_bf16(float f) {
  unsigned int u = __float_as_uint(f);
  unsigned int r = u + 0x7FFFu + ((u >> 16) & 1u);   // round-to-nearest-even
  return (unsigned short)(r >> 16);
}

// ---------------- TDM: 2D bf16 tile -> LDS (padded rows) ----------------
// Stages tile_d1 rows x tile_d0 elements (2B each) starting at gptr with a row
// stride of stride0 elements. LDS destination gets 4 DWORDs (16B) of padding
// after every 16 DWORDs (64B row) => 80B LDS row stride (bank-conflict free,
// 16B aligned for ds_load_b128). D# packing per CDNA5 ISA 8.3/8.4.
__device__ __forceinline__ void tdm_load_tile_bf16(
    unsigned lds_addr, const void* gptr,
    unsigned long long stride0_elems,
    unsigned tile_d0, unsigned tile_d1) {
  unsigned long long ga = (unsigned long long)gptr;
  v4u g0;
  g0[0] = 1u;                                         // count=1, user mode
  g0[1] = lds_addr;                                   // LDS byte address
  g0[2] = (unsigned)(ga & 0xFFFFFFFFu);               // global_addr[31:0]
  g0[3] = (unsigned)((ga >> 32) & 0x1FFFFFFu)         // global_addr[56:32]
        | (2u << 30);                                 // type=2 ("image")
  const unsigned td0 = 1u << 20, td1 = 1u << 20;      // tensor dims >> tile (no OOB)
  v8i_t g1;
  g1[0] = (1 << 16)                                   // data_size = 1 (2 bytes)
        | (1 << 20)                                   // pad_enable
        | (3 << 22)                                   // pad_interval: 16 DWORDs
        | (3 << 25);                                  // pad_amount: 4 DWORDs
  g1[1] = (int)((td0 & 0xFFFFu) << 16);               // tensor_dim0[15:0]
  g1[2] = (int)(((td0 >> 16) & 0xFFFFu) | ((td1 & 0xFFFFu) << 16));
  g1[3] = (int)(((td1 >> 16) & 0xFFFFu) | (tile_d0 << 16));
  g1[4] = (int)(tile_d1 & 0xFFFFu);                   // tile_dim1 (tile_dim2=0)
  g1[5] = (int)(stride0_elems & 0xFFFFFFFFull);       // dim0 stride low
  g1[6] = (int)((stride0_elems >> 32) & 0xFFFFull);   // dim0 stride high
  g1[7] = 0;
  v4i_t z4 = {0, 0, 0, 0};                            // groups 2/3 unused (2D)
  v8i_t z8 = {0, 0, 0, 0, 0, 0, 0, 0};
  __builtin_amdgcn_tensor_load_to_lds(g0, g1, z4, z4, z8, 0);  // 6-arg form
}

// ---------------- TDM-staged WMMA GEMM: C(MxN) = A(MxK) * Bt(NxK)^T ---------
// Block: 256 threads (8 waves) computes a 128x128 C tile; wave (2x4 grid)
// computes 64Mx32N (8 accumulators). K-step 32, double-buffered TDM staging.
// Requires M%128==0, N%128==0, K%32==0.
#define LDS_ROW_USH  40     // 80B padded LDS row stride (32 bf16 data + 16B pad)
#define STAGE_USH    (128 * LDS_ROW_USH)        // one A or B stage: 10240 B
#define BUF_USH      (2 * STAGE_USH)            // A + B per buffer

__global__ __launch_bounds__(256)
void wmma_gemm_tdm_bf16(const unsigned short* __restrict__ A,
                        const unsigned short* __restrict__ Bt,
                        float* __restrict__ C,
                        int M, int N, int Kd) {
  __shared__ alignas(16) unsigned short smem[2 * BUF_USH];   // 40 KB

  const int lane  = threadIdx.x & 31;
  const int wave  = threadIdx.x >> 5;
  const int mr    = lane & 15;
  const int khalf = lane >> 4;
  const int wm    = wave >> 2;          // 0..1  -> 64-row slab
  const int wn    = wave & 3;           // 0..3  -> 32-col slab
  const int m0    = blockIdx.y * 128;
  const int n0    = blockIdx.x * 128;

  const unsigned lds_base = (unsigned)(unsigned long long)(uintptr_t)smem;
  const int nk = Kd / 32;

  v8f acc[4][2];
#pragma unroll
  for (int sm = 0; sm < 4; ++sm)
#pragma unroll
    for (int sn = 0; sn < 2; ++sn)
      acc[sm][sn] = {};

  // prologue: stage K-tile 0 into buffer 0 (one DMA pair per workgroup)
  if (wave == 0) {
    tdm_load_tile_bf16(lds_base,             A  + (size_t)m0 * Kd, Kd, 32, 128);
    tdm_load_tile_bf16(lds_base + STAGE_USH * 2, Bt + (size_t)n0 * Kd, Kd, 32, 128);
  }

  for (int ki = 0; ki < nk; ++ki) {
    const int cur = ki & 1;
    if (wave == 0) __builtin_amdgcn_s_wait_tensorcnt(0);
    __syncthreads();                                   // tile ki visible to all

    if (wave == 0 && ki + 1 < nk) {                    // overlap next DMA w/ math
      const unsigned nxt = (unsigned)((ki + 1) & 1);
      const int kk = (ki + 1) * 32;
      tdm_load_tile_bf16(lds_base + nxt * (2u * BUF_USH),
                         A + (size_t)m0 * Kd + kk, Kd, 32, 128);
      tdm_load_tile_bf16(lds_base + nxt * (2u * BUF_USH) + STAGE_USH * 2,
                         Bt + (size_t)n0 * Kd + kk, Kd, 32, 128);
    }

    const unsigned short* sA = smem + (size_t)cur * BUF_USH;
    const unsigned short* sB = sA + STAGE_USH;

    Frag a[4], b[2];
#pragma unroll
    for (int sm = 0; sm < 4; ++sm) {
      const unsigned short* r = sA + (wm * 64 + sm * 16 + mr) * LDS_ROW_USH;
      a[sm].u[0] = *(const U4*)(r + khalf * 8);        // K = khalf*8 .. +7
      a[sm].u[1] = *(const U4*)(r + 16 + khalf * 8);   // K = 16+khalf*8 .. +7
    }
#pragma unroll
    for (int sn = 0; sn < 2; ++sn) {
      const unsigned short* r = sB + (wn * 32 + sn * 16 + mr) * LDS_ROW_USH;
      b[sn].u[0] = *(const U4*)(r + khalf * 8);        // K = khalf*8 .. +7
      b[sn].u[1] = *(const U4*)(r + 16 + khalf * 8);   // K = 16+khalf*8 .. +7
    }
#pragma unroll
    for (int sm = 0; sm < 4; ++sm)
#pragma unroll
      for (int sn = 0; sn < 2; ++sn)
        acc[sm][sn] = __builtin_amdgcn_wmma_f32_16x16x32_bf16(
            false, a[sm].v, false, b[sn].v, (short)0, acc[sm][sn], false, false);

    __syncthreads();                                   // done reading buf cur
  }

  // epilogue: VGPR r -> row (.. + 8*khalf + r), col (.. + mr)
#pragma unroll
  for (int sm = 0; sm < 4; ++sm)
#pragma unroll
    for (int sn = 0; sn < 2; ++sn) {
      float* crow = C + (size_t)(m0 + wm * 64 + sm * 16 + khalf * 8) * N
                      + (n0 + wn * 32 + sn * 16 + mr);
#pragma unroll
      for (int r = 0; r < 8; ++r)
        crow[(size_t)r * N] = acc[sm][sn][r];
    }
}

// NOTE on buffer addressing above: buffer i occupies smem[i*BUF_USH ..], i.e.
// byte offset i*2*BUF_USH -- we use ushort units for smem indexing and
// BYTE units for the TDM lds_addr, hence the factor-2 on STAGE_USH/BUF_USH in
// the TDM calls (STAGE_USH ushorts == STAGE_USH*2 bytes).

// ---------------- fallback direct-global WMMA GEMM (skinny N) ----------------
__global__ __launch_bounds__(256)
void wmma_gemm_bf16(const unsigned short* __restrict__ A,
                    const unsigned short* __restrict__ Bt,
                    float* __restrict__ C,
                    int M, int N, int Kd) {
  const int lane  = threadIdx.x & 31;
  const int wave  = threadIdx.x >> 5;
  const int mr    = lane & 15;
  const int khalf = lane >> 4;
  const int n0 = (blockIdx.x * 8 + wave) * 16;
  const int m0 = blockIdx.y * 16;
  if (n0 >= N) return;                        // whole-wave uniform exit

  const unsigned short* arow = A  + (size_t)(m0 + mr) * Kd;
  const unsigned short* brow = Bt + (size_t)(n0 + mr) * Kd;

  v8f acc = {};
  for (int kk = 0; kk < Kd; kk += 32) {
    Frag a, b;
    a.u[0] = *(const U4*)(arow + kk + khalf * 8);
    a.u[1] = *(const U4*)(arow + kk + 16 + khalf * 8);
    b.u[0] = *(const U4*)(brow + kk + khalf * 8);
    b.u[1] = *(const U4*)(brow + kk + 16 + khalf * 8);
    acc = __builtin_amdgcn_wmma_f32_16x16x32_bf16(false, a.v, false, b.v,
                                                  (short)0, acc, false, false);
  }
  float* crow = C + (size_t)(m0 + khalf * 8) * N + (n0 + mr);
#pragma unroll
  for (int r = 0; r < 8; ++r)
    crow[(size_t)r * N] = acc[r];
}

// ---------------- elementwise converters ----------------
__global__ void convert_bf16_kernel(const float* __restrict__ in,
                                    unsigned short* __restrict__ out, int n) {
  int i = blockIdx.x * blockDim.x + threadIdx.x;
  if (i < n) out[i] = f32_to_bf16(in[i]);
}

// in: R x Cc (f32 row-major)  ->  out: Cc x R (bf16 row-major)
__global__ void transpose_bf16_kernel(const float* __restrict__ in,
                                      unsigned short* __restrict__ out,
                                      int R, int Cc) {
  int i = blockIdx.x * blockDim.x + threadIdx.x;
  if (i >= R * Cc) return;
  int r = i / Cc, c = i % Cc;
  out[(size_t)c * R + r] = f32_to_bf16(in[(size_t)r * Cc + c]);
}

// ---------------- depthwise conv(K=4) + SiLU ----------------
__global__ void conv_silu_kernel(const float* __restrict__ xz,
                                 const float* __restrict__ conv_w,
                                 const float* __restrict__ conv_b,
                                 float* __restrict__ xc,
                                 unsigned short* __restrict__ xc_bf) {
  int idx = blockIdx.x * blockDim.x + threadIdx.x;   // over NTOK*D_INNER
  if (idx >= NTOK * D_INNER) return;
  int d = idx % D_INNER;
  int t = idx / D_INNER;
  int l = t % SEQ;
  float acc = conv_b[d];
#pragma unroll
  for (int k = 0; k < CONV_K; ++k) {
    int ls = l - (CONV_K - 1) + k;
    if (ls >= 0)
      acc += xz[(size_t)(t - l + ls) * XZ_N + d] * conv_w[d * CONV_K + k];
  }
  float s = acc / (1.f + __expf(-acc));              // SiLU
  xc[idx] = s;
  xc_bf[idx] = f32_to_bf16(s);
}

// ---------------- extract dt-rank slice of x_dbl as bf16 ----------------
__global__ void extract_dtin_kernel(const float* __restrict__ xdbl,
                                    unsigned short* __restrict__ dtin_bf) {
  int i = blockIdx.x * blockDim.x + threadIdx.x;     // over NTOK*DT_RANK
  if (i >= NTOK * DT_RANK) return;
  int t = i / DT_RANK, j = i % DT_RANK;
  dtin_bf[i] = f32_to_bf16(xdbl[(size_t)t * XDBL_N + j]);
}

// ---------------- dt = softplus(dt_raw + b_dt) ----------------
__global__ void softplus_bias_kernel(float* __restrict__ dt,
                                     const float* __restrict__ b_dt) {
  int i = blockIdx.x * blockDim.x + threadIdx.x;     // over NTOK*D_INNER
  if (i >= NTOK * D_INNER) return;
  float v = dt[i] + b_dt[i % D_INNER];
  dt[i] = (v > 20.f) ? v : log1pf(__expf(v));
}

// ---------------- selective scan: one thread per (batch, channel) ----------------
__global__ void selective_scan_kernel(const float* __restrict__ dt,
                                      const float* __restrict__ xc,
                                      const float* __restrict__ xz,   // z half
                                      const float* __restrict__ xdbl, // B/C rows
                                      const float* __restrict__ A_log,
                                      const float* __restrict__ Dvec,
                                      unsigned short* __restrict__ y_bf) {
  int idx = blockIdx.x * blockDim.x + threadIdx.x;   // over BATCH*D_INNER
  if (idx >= BATCH * D_INNER) return;
  int d = idx % D_INNER;
  int b = idx / D_INNER;

  float A[D_STATE], h[D_STATE];
#pragma unroll
  for (int n = 0; n < D_STATE; ++n) {
    A[n] = -__expf(A_log[d * D_STATE + n]);
    h[n] = 0.f;
  }
  const float Dd = Dvec[d];

  for (int l = 0; l < SEQ; ++l) {
    size_t t = (size_t)b * SEQ + l;
    float dtv = dt[t * D_INNER + d];
    float u   = xc[t * D_INNER + d];
    float zv  = xz[t * XZ_N + D_INNER + d];
    const float* Bm = xdbl + t * XDBL_N + DT_RANK;   // same row across lanes
    const float* Cm = Bm + D_STATE;
    float dtu = dtv * u;
    float y = 0.f;
#pragma unroll
    for (int n = 0; n < D_STATE; ++n) {
      float dA = __expf(dtv * A[n]);                 // v_exp_f32
      h[n] = dA * h[n] + dtu * Bm[n];
      y += h[n] * Cm[n];
    }
    y += Dd * u;
    float sz = zv / (1.f + __expf(-zv));             // SiLU gate
    y_bf[t * D_INNER + d] = f32_to_bf16(y * sz);
  }
}

// ---------------- host-side launch ----------------
extern "C" void kernel_launch(void* const* d_in, const int* in_sizes, int n_in,
                              void* d_out, int out_size, void* d_ws, size_t ws_size,
                              hipStream_t stream) {
  const float* x      = (const float*)d_in[0];
  const float* W_in   = (const float*)d_in[1];  // D_MODEL x 2*D_INNER
  const float* conv_w = (const float*)d_in[2];  // D_INNER x 4
  const float* conv_b = (const float*)d_in[3];
  const float* W_x    = (const float*)d_in[4];  // D_INNER x 96
  const float* W_dt   = (const float*)d_in[5];  // DT_RANK x D_INNER
  const float* b_dt   = (const float*)d_in[6];
  const float* A_log  = (const float*)d_in[7];
  const float* Dvec   = (const float*)d_in[8];
  const float* W_out  = (const float*)d_in[9];  // D_INNER x D_MODEL
  float* out = (float*)d_out;

  // workspace carve-up (all chunks 256B-multiples; hipMalloc base 256B aligned)
  char* p = (char*)d_ws;
  float* xz   = (float*)p;  p += (size_t)NTOK * XZ_N * 4;
  float* xc   = (float*)p;  p += (size_t)NTOK * D_INNER * 4;
  float* dt   = (float*)p;  p += (size_t)NTOK * D_INNER * 4;
  float* xdbl = (float*)p;  p += (size_t)NTOK * XDBL_N * 4;
  unsigned short* x_bf    = (unsigned short*)p; p += (size_t)NTOK * D_MODEL * 2;
  unsigned short* Win_t   = (unsigned short*)p; p += (size_t)XZ_N * D_MODEL * 2;
  unsigned short* xc_bf   = (unsigned short*)p; p += (size_t)NTOK * D_INNER * 2;
  unsigned short* Wx_t    = (unsigned short*)p; p += (size_t)XDBL_N * D_INNER * 2;
  unsigned short* dtin_bf = (unsigned short*)p; p += (size_t)NTOK * DT_RANK * 2;
  unsigned short* Wdt_t   = (unsigned short*)p; p += (size_t)D_INNER * DT_RANK * 2;
  unsigned short* y_bf    = (unsigned short*)p; p += (size_t)NTOK * D_INNER * 2;
  unsigned short* Wout_t  = (unsigned short*)p; p += (size_t)D_MODEL * D_INNER * 2;

  const int T = 256;
  auto cdiv = [](int a, int b) { return (a + b - 1) / b; };

  // 1) x -> bf16 ; W_in -> bf16 transposed (N x K)
  convert_bf16_kernel<<<cdiv(NTOK * D_MODEL, T), T, 0, stream>>>(x, x_bf, NTOK * D_MODEL);
  transpose_bf16_kernel<<<cdiv(D_MODEL * XZ_N, T), T, 0, stream>>>(W_in, Win_t, D_MODEL, XZ_N);

  // 2) xz = x @ W_in   (4096 x 4096, K=1024)  [TDM + WMMA]
  wmma_gemm_tdm_bf16<<<dim3(XZ_N / 128, NTOK / 128), T, 0, stream>>>(x_bf, Win_t, xz,
                                                                     NTOK, XZ_N, D_MODEL);

  // 3) depthwise conv + SiLU -> xc (f32 + bf16)
  conv_silu_kernel<<<cdiv(NTOK * D_INNER, T), T, 0, stream>>>(xz, conv_w, conv_b, xc, xc_bf);

  // 4) x_dbl = xc @ W_x   (4096 x 96, K=2048)  [skinny: direct-global WMMA]
  transpose_bf16_kernel<<<cdiv(D_INNER * XDBL_N, T), T, 0, stream>>>(W_x, Wx_t, D_INNER, XDBL_N);
  wmma_gemm_bf16<<<dim3(cdiv(XDBL_N, 128), NTOK / 16), T, 0, stream>>>(xc_bf, Wx_t, xdbl,
                                                                       NTOK, XDBL_N, D_INNER);

  // 5) dt = softplus(x_dbl[:, :64] @ W_dt + b_dt)   (4096 x 2048, K=64) [TDM + WMMA]
  extract_dtin_kernel<<<cdiv(NTOK * DT_RANK, T), T, 0, stream>>>(xdbl, dtin_bf);
  transpose_bf16_kernel<<<cdiv(DT_RANK * D_INNER, T), T, 0, stream>>>(W_dt, Wdt_t, DT_RANK, D_INNER);
  wmma_gemm_tdm_bf16<<<dim3(D_INNER / 128, NTOK / 128), T, 0, stream>>>(dtin_bf, Wdt_t, dt,
                                                                        NTOK, D_INNER, DT_RANK);
  softplus_bias_kernel<<<cdiv(NTOK * D_INNER, T), T, 0, stream>>>(dt, b_dt);

  // 6) selective scan + skip + SiLU(z) gate -> y (bf16)
  selective_scan_kernel<<<cdiv(BATCH * D_INNER, T), T, 0, stream>>>(dt, xc, xz, xdbl,
                                                                    A_log, Dvec, y_bf);

  // 7) out = y @ W_out   (4096 x 1024, K=2048)  [TDM + WMMA]
  transpose_bf16_kernel<<<cdiv(D_INNER * D_MODEL, T), T, 0, stream>>>(W_out, Wout_t, D_INNER, D_MODEL);
  wmma_gemm_tdm_bf16<<<dim3(D_MODEL / 128, NTOK / 128), T, 0, stream>>>(y_bf, Wout_t, out,
                                                                        NTOK, D_MODEL, D_INNER);
}
